// OR_LSTM_35785667510924
// MI455X (gfx1250) — matrix operand order repaired and checked by reading, not compiled, not run on hardware
//
#include <hip/hip_runtime.h>
#include <hip/hip_bf16.h>

typedef __attribute__((ext_vector_type(8)))  float  v8f;
typedef __attribute__((ext_vector_type(16))) __bf16 v16bf;

namespace {

constexpr int kB    = 64;
constexpr int kT    = 256;
constexpr int kTout = 248;                    // T - WS
constexpr int kHnOff = kB * kTout * 2;        // 31744
constexpr int kCnOff = kHnOff + 2 * kB * 64;  // 39936

union FragU { v16bf v; uint4 u[2]; };

// A-matrix (16-bit 16x32): lane half -> interleaved K groups of 8 (ISA 7.12.2)
__device__ __forceinline__ v16bf load_fragA(const __bf16* rowbase, int kc, int hl) {
    const __bf16* p = rowbase + kc * 32 + hl * 8;
    FragU f;
    f.u[0] = *reinterpret_cast<const uint4*>(p);       // K = kc*32 + hl*8 + 0..7
    f.u[1] = *reinterpret_cast<const uint4*>(p + 16);  // K = kc*32 + 16 + hl*8 + 0..7
    return f.v;
}
// B-matrix (16-bit 32x16): lane half -> interleaved K groups of 8 (matches A)
__device__ __forceinline__ v16bf load_fragB(const __bf16* rowbase, int kc, int hl) {
    const __bf16* p = rowbase + kc * 32 + hl * 8;
    FragU f;
    f.u[0] = *reinterpret_cast<const uint4*>(p);
    f.u[1] = *reinterpret_cast<const uint4*>(p + 16);
    return f.v;
}

#define WMMA_BF16(A, Bv, C) \
    __builtin_amdgcn_wmma_f32_16x16x32_bf16(false, (A), false, (Bv), (short)0, (C), false, false)

__device__ __forceinline__ float sigm(float x) { return 1.0f / (1.0f + __expf(-x)); }
__device__ __forceinline__ float tanh_fast(float x) {
    x = fminf(fmaxf(x, -15.0f), 15.0f);
    float e = __expf(2.0f * x);
    return (e - 1.0f) / (e + 1.0f);
}

// Shared gate-fusion tail: D-tile element (r) -> row m = mtile*16 + hl*8 + r, col = j*16 + nl
__device__ __forceinline__ void gate_update(const v8f acc[4], float* c,
                                            __bf16* Hout, float* HoutF,
                                            int mtile, int j, int hl, int nl) {
    const int col = j * 16 + nl;
#pragma unroll
    for (int r = 0; r < 8; ++r) {
        const int m   = mtile * 16 + hl * 8 + r;
        const float ig = sigm(acc[0][r]);
        const float fg = sigm(acc[1][r]);
        const float gg = tanh_fast(acc[2][r]);
        const float og = sigm(acc[3][r]);
        const float cn = fg * c[r] + ig * gg;
        c[r] = cn;
        const float h = og * tanh_fast(cn);
        Hout[m * 64 + col]  = (__bf16)h;
        HoutF[m * 64 + col] = h;
    }
    __syncthreads();  // h visible to all waves before next consumer
}

// Layer 0: K_in = 4 done with VALU FMAs; recurrent 64-K path with WMMA bf16
__device__ __forceinline__ void cell_l0(const float* Xf, const float* Wih0f,
                                        const __bf16* Hb, const __bf16* Whh,
                                        const float* bias, float* c,
                                        __bf16* Hout, float* HoutF,
                                        int mtile, int j, int hl, int nl) {
    const int mrow = mtile * 16 + nl;
    v16bf ah0 = load_fragA(Hb + mrow * 64, 0, hl);
    v16bf ah1 = load_fragA(Hb + mrow * 64, 1, hl);
    float4 xr[8];
#pragma unroll
    for (int r = 0; r < 8; ++r) {
        const int m = mtile * 16 + hl * 8 + r;
        xr[r] = *reinterpret_cast<const float4*>(Xf + m * 4);
    }
    __syncthreads();  // finish reading old H/X before anyone overwrites
    v8f acc[4];
#pragma unroll
    for (int gi = 0; gi < 4; ++gi) {
        const int grow = gi * 64 + j * 16 + nl;
        const float4 wv = *reinterpret_cast<const float4*>(Wih0f + grow * 4);
        const float  bv = bias[grow];
        v8f a;
#pragma unroll
        for (int r = 0; r < 8; ++r)
            a[r] = bv + xr[r].x * wv.x + xr[r].y * wv.y + xr[r].z * wv.z + xr[r].w * wv.w;
        a = WMMA_BF16(ah0, load_fragB(Whh + grow * 64, 0, hl), a);
        a = WMMA_BF16(ah1, load_fragB(Whh + grow * 64, 1, hl), a);
        acc[gi] = a;
    }
    gate_update(acc, c, Hout, HoutF, mtile, j, hl, nl);
}

// Layer 1: both input (h0) and recurrent paths via WMMA bf16 (K = 64 each)
__device__ __forceinline__ void cell_l1(const __bf16* Xb, const __bf16* Wih,
                                        const __bf16* Hb, const __bf16* Whh,
                                        const float* bias, float* c,
                                        __bf16* Hout, float* HoutF,
                                        int mtile, int j, int hl, int nl) {
    const int mrow = mtile * 16 + nl;
    v16bf ax0 = load_fragA(Xb + mrow * 64, 0, hl);
    v16bf ax1 = load_fragA(Xb + mrow * 64, 1, hl);
    v16bf ah0 = load_fragA(Hb + mrow * 64, 0, hl);
    v16bf ah1 = load_fragA(Hb + mrow * 64, 1, hl);
    __syncthreads();
    v8f acc[4];
#pragma unroll
    for (int gi = 0; gi < 4; ++gi) {
        const int grow = gi * 64 + j * 16 + nl;
        const float bv = bias[grow];
        v8f a;
#pragma unroll
        for (int r = 0; r < 8; ++r) a[r] = bv;
        a = WMMA_BF16(ax0, load_fragB(Wih + grow * 64, 0, hl), a);
        a = WMMA_BF16(ax1, load_fragB(Wih + grow * 64, 1, hl), a);
        a = WMMA_BF16(ah0, load_fragB(Whh + grow * 64, 0, hl), a);
        a = WMMA_BF16(ah1, load_fragB(Whh + grow * 64, 1, hl), a);
        acc[gi] = a;
    }
    gate_update(acc, c, Hout, HoutF, mtile, j, hl, nl);
}

} // namespace

__global__ void __launch_bounds__(512, 1) or_lstm_persistent(
    const float* __restrict__ traj,
    const float* __restrict__ Wih0, const float* __restrict__ Whh0,
    const float* __restrict__ bih0, const float* __restrict__ bhh0,
    const float* __restrict__ Wih1, const float* __restrict__ Whh1,
    const float* __restrict__ bih1, const float* __restrict__ bhh1,
    const float* __restrict__ Wlin, const float* __restrict__ blin,
    float* __restrict__ out)
{
    // ---- LDS (~157 KB, under 160 KB CU-mode addressable) ----
    __shared__ __align__(16) __bf16 sWhh0[256 * 64];
    __shared__ __align__(16) __bf16 sWih1[256 * 64];
    __shared__ __align__(16) __bf16 sWhh1[256 * 64];
    __shared__ __align__(16) float  sWih0f[256 * 4];
    __shared__ float sB0[256], sB1[256];
    __shared__ float sWlin[128], sBlin[2];
    __shared__ __align__(16) float  sXf[64 * 4];
    __shared__ __align__(16) __bf16 sH0[64 * 64], sH1[64 * 64];
    __shared__ __align__(16) float  sH0f[64 * 64], sH1f[64 * 64];
    __shared__ float sBuf[64 * 8 * 2];   // sliding prediction window
    __shared__ float sLin[128];

    const int tid   = (int)threadIdx.x;
    const int lane  = tid & 31;
    const int w     = tid >> 5;       // 16 waves
    const int nl    = lane & 15;
    const int hl    = (lane >> 4) & 1;
    const int mtile = w & 3;          // which 16 batch rows
    const int j     = w >> 2;         // which 16 hidden columns (per gate)

    // ---- stage weights once (bf16 for WMMA, f32 biases) ----
    for (int i = tid; i < 256 * 64; i += 512) {
        sWhh0[i] = (__bf16)Whh0[i];
        sWih1[i] = (__bf16)Wih1[i];
        sWhh1[i] = (__bf16)Whh1[i];
    }
    for (int i = tid; i < 256 * 4; i += 512) sWih0f[i] = Wih0[i];
    for (int i = tid; i < 256; i += 512) { sB0[i] = bih0[i] + bhh0[i]; sB1[i] = bih1[i] + bhh1[i]; }
    if (tid < 128) sWlin[tid] = Wlin[tid];
    if (tid < 2)   sBlin[tid] = blin[tid];
    __syncthreads();

    float c0[8], c1[8];  // persistent cell-state fragments (register-resident)

    // 248 window calls: ci==0 warmup, 1..7 feedback warmup, 8..247 scan
    for (int ci = 0; ci < 248; ++ci) {
        const int mode = (ci == 0) ? 0 : (ci < 8 ? 1 : 2);
        const int t = ci;

#pragma unroll
        for (int r = 0; r < 8; ++r) { c0[r] = 0.0f; c1[r] = 0.0f; }
        for (int i = tid; i < 64 * 64; i += 512) { sH0[i] = (__bf16)0.0f; sH1[i] = (__bf16)0.0f; }

        for (int s = 0; s < 8; ++s) {
            if (tid < 256) {  // assemble x_t [64,4]
                const int b = tid >> 2, ch = tid & 3;
                float val;
                if (mode == 0) {
                    val = traj[(b * kT + s) * 4 + ch];
                } else if (mode == 1) {
                    if (s < 8 - t) {
                        val = traj[(b * kT + t + s) * 4 + ch];
                    } else {
                        const int k = s - (8 - t);
                        val = (ch < 2) ? traj[(b * kT + 7 + t + k) * 4 + ch]
                                       : sBuf[(b * 8 + k) * 2 + (ch - 2)];
                    }
                } else {
                    val = (ch < 2) ? traj[(b * kT + t + s) * 4 + ch]
                                   : sBuf[(b * 8 + s) * 2 + (ch - 2)];
                }
                sXf[b * 4 + ch] = val;
            }
            __syncthreads();

            cell_l0(sXf, sWih0f, sH0, sWhh0, sB0, c0, sH0, sH0f, mtile, j, hl, nl);
            cell_l1(sH0, sWih1, sH1, sWhh1, sB1, c1, sH1, sH1f, mtile, j, hl, nl);
        }

        // lin(h1_last): [64,64] @ [64,2] + b
        if (tid < 128) {
            const int b = tid >> 1, o = tid & 1;
            float a = sBlin[o];
#pragma unroll 8
            for (int k = 0; k < 64; ++k) a += sH1f[b * 64 + k] * sWlin[o * 64 + k];
            sLin[tid] = a;
        }
        __syncthreads();

        if (mode == 2) {  // shift buffer + append prediction
            float sh[8];
            const int b = tid >> 1, o = tid & 1;
            if (tid < 128) {
#pragma unroll
                for (int r = 0; r < 7; ++r) sh[r] = sBuf[(b * 8 + r + 1) * 2 + o];
                sh[7] = sBuf[(b * 8 + 7) * 2 + o] + sLin[tid];
            }
            __syncthreads();
            if (tid < 128) {
#pragma unroll
                for (int r = 0; r < 8; ++r) sBuf[(b * 8 + r) * 2 + o] = sh[r];
                out[(b * kTout + t) * 2 + o] = sh[7];
            }
        } else {          // warmup: grow buffer
            if (tid < 128) {
                const int b = tid >> 1, o = tid & 1;
                const float prev = (mode == 0) ? traj[(b * kT + 7) * 4 + 2 + o]
                                               : sBuf[(b * 8 + t - 1) * 2 + o];
                const float nv = prev + sLin[tid];
                sBuf[(b * 8 + t) * 2 + o] = nv;
                out[(b * kTout + t) * 2 + o] = nv;
            }
        }
        __syncthreads();
    }

    // final hidden states: h_n [2,64,64] from f32 LDS, c_n [2,64,64] from register frags
    for (int i = tid; i < 64 * 64; i += 512) {
        out[kHnOff + i]        = sH0f[i];
        out[kHnOff + 4096 + i] = sH1f[i];
    }
    {
        const int col = j * 16 + nl;
#pragma unroll
        for (int r = 0; r < 8; ++r) {
            const int m = mtile * 16 + hl * 8 + r;
            out[kCnOff + m * 64 + col]        = c0[r];
            out[kCnOff + 4096 + m * 64 + col] = c1[r];
        }
    }
}

extern "C" void kernel_launch(void* const* d_in, const int* in_sizes, int n_in,
                              void* d_out, int out_size, void* d_ws, size_t ws_size,
                              hipStream_t stream) {
    (void)in_sizes; (void)n_in; (void)out_size; (void)d_ws; (void)ws_size;
    const float* traj = (const float*)d_in[0];
    const float* Wih0 = (const float*)d_in[1];
    const float* Whh0 = (const float*)d_in[2];
    const float* bih0 = (const float*)d_in[3];
    const float* bhh0 = (const float*)d_in[4];
    const float* Wih1 = (const float*)d_in[5];
    const float* Whh1 = (const float*)d_in[6];
    const float* bih1 = (const float*)d_in[7];
    const float* bhh1 = (const float*)d_in[8];
    const float* Wlin = (const float*)d_in[9];
    const float* blin = (const float*)d_in[10];
    or_lstm_persistent<<<dim3(1), dim3(512), 0, stream>>>(
        traj, Wih0, Whh0, bih0, bhh0, Wih1, Whh1, bih1, bhh1, Wlin, blin,
        (float*)d_out);
}
